// spiking_coESN_66915590471999
// MI455X (gfx1250) — matrix operand order repaired and checked by reading, not compiled, not run on hardware
//
#include <hip/hip_runtime.h>

// ---------------------------------------------------------------------------
// Spiking coupled-ESN on MI455X (gfx1250), persistent-kernel design.
//   grid = 16 WGs x 256 threads (8 waves). WG b owns hidden cols [128b,128b+128)
//   for all 32 batches. Per step:
//     - async-copy spike matrix s[32x2048] (bf16, 128 KB) global -> LDS
//       (global_load_async_to_lds_b128, overlapped with x-projection WMMA),
//     - C[32x128] = xt@x2h + s@h2h via v_wmma_f32_16x16x32_bf16
//       (A from LDS ds_load_b128, B streamed from L2-resident h2h),
//     - fp32 neuron update in registers, spikes ping-ponged through global,
//     - one atomic grid barrier per step.
// ---------------------------------------------------------------------------

typedef __bf16 bf16;
typedef __attribute__((ext_vector_type(16))) __bf16 v16bf;
typedef __attribute__((ext_vector_type(8)))  __bf16 v8bf;
typedef __attribute__((ext_vector_type(8)))  float  v8f;

// async-to-LDS builtin operand types (probe: arg0 is AS(1) int __vector(4)*)
typedef int v4i __attribute__((vector_size(16)));
typedef __attribute__((address_space(1))) v4i* as1_v4i;
typedef __attribute__((address_space(3))) v4i* as3_v4i;

#define N_INP   96
#define N_HID   2048
#define BATCH   32
#define TSTEPS  1024
#define NBLOCKS 16
#define SLICE   (N_HID / NBLOCKS)   // 128 hidden cols per WG
#define LDS_STRIDE 2056             // 2048 + 8 bf16 (16B row skew -> no bank conflicts)

// dynamics constants (dt = 0.075)
#define DT          0.075f
#define LIF_KEEP    0.85f            // 1 - dt/tau_m, tau_m = 0.5
#define GAMMA       2.7f
#define EPSI        4.7f
#define SPIKE_GAIN  5.0f
#define REF_DECAY   0.74081822066817179f   // exp(-dt/0.25)

#if __has_builtin(__builtin_amdgcn_global_load_async_to_lds_b128)
#define HAVE_ASYNC_LDS 1
#else
#define HAVE_ASYNC_LDS 0
#endif

// ---------------------------------------------------------------------------
// prep kernels
// ---------------------------------------------------------------------------
__global__ void f32_to_bf16_kernel(const float* __restrict__ in,
                                   bf16* __restrict__ out, int n) {
    int i = blockIdx.x * blockDim.x + threadIdx.x;
    if (i < n) out[i] = (bf16)in[i];
}

// x: [BATCH][TSTEPS][N_INP] fp32  ->  xbf: [TSTEPS][BATCH][N_INP] bf16
__global__ void x_time_major_bf16_kernel(const float* __restrict__ x,
                                         bf16* __restrict__ out) {
    int i = blockIdx.x * blockDim.x + threadIdx.x;
    if (i >= BATCH * TSTEPS * N_INP) return;
    int k = i % N_INP;
    int t = (i / N_INP) % TSTEPS;
    int m = i / (N_INP * TSTEPS);
    out[(size_t)t * (BATCH * N_INP) + m * N_INP + k] = (bf16)x[i];
}

// ---------------------------------------------------------------------------
// A-tile gather (16x32 bf16, row-major source, row stride in elements):
// lane holds row m, chunks K = g*8..g*8+7 and 16+g*8..16+g*8+7 (ISA §7.12.2)
// ---------------------------------------------------------------------------
__device__ __forceinline__ v16bf load_a16(const bf16* base, int row_stride,
                                          int m, int kbase, int g) {
    const bf16* p0 = base + (size_t)m * row_stride + kbase + g * 8;
    v8bf lo = *(const v8bf*)p0;          // 16B
    v8bf hi = *(const v8bf*)(p0 + 16);   // 16B
    return __builtin_shufflevector(lo, hi, 0,1,2,3,4,5,6,7,8,9,10,11,12,13,14,15);
}

// ---------------------------------------------------------------------------
// async copy of full spike matrix [32][2048] bf16 (global) -> LDS (skewed rows)
// each of 256 threads copies one 16B chunk per row (256 chunks/row, 32 rows)
// ---------------------------------------------------------------------------
__device__ __forceinline__ void issue_spike_copy(const bf16* __restrict__ src,
                                                 bf16* dst_lds, int tid) {
#if HAVE_ASYNC_LDS
#pragma unroll
    for (int r = 0; r < BATCH; ++r) {
        const bf16* g = src + (size_t)r * N_HID + tid * 8;
        bf16*       l = dst_lds + (size_t)r * LDS_STRIDE + tid * 8;
        __builtin_amdgcn_global_load_async_to_lds_b128((as1_v4i)g, (as3_v4i)l, 0, 0);
    }
#else
#pragma unroll
    for (int r = 0; r < BATCH; ++r) {
        *(v8bf*)(dst_lds + (size_t)r * LDS_STRIDE + tid * 8) =
            *(const v8bf*)(src + (size_t)r * N_HID + tid * 8);
    }
#endif
}

__device__ __forceinline__ void wait_spike_copy() {
#if HAVE_ASYNC_LDS
#if __has_builtin(__builtin_amdgcn_s_wait_asynccnt)
    __builtin_amdgcn_s_wait_asynccnt(0);
#else
    asm volatile("s_wait_asynccnt 0x0" ::: "memory");
#endif
#endif
    __syncthreads();
}

// ---------------------------------------------------------------------------
// sense-reversal grid barrier (16 co-resident blocks)
// ---------------------------------------------------------------------------
__device__ __forceinline__ void grid_barrier(unsigned* ctrl) {
    __threadfence();          // make this thread's spike stores device-visible
    __syncthreads();          // also: all waves done reading LDS spike tile
    if (threadIdx.x == 0) {
        unsigned gen = __hip_atomic_load(&ctrl[1], __ATOMIC_RELAXED,
                                         __HIP_MEMORY_SCOPE_AGENT);
        unsigned prev = __hip_atomic_fetch_add(&ctrl[0], 1u, __ATOMIC_ACQ_REL,
                                               __HIP_MEMORY_SCOPE_AGENT);
        if (prev == NBLOCKS - 1) {
            __hip_atomic_store(&ctrl[0], 0u, __ATOMIC_RELAXED,
                               __HIP_MEMORY_SCOPE_AGENT);
            __hip_atomic_fetch_add(&ctrl[1], 1u, __ATOMIC_RELEASE,
                                   __HIP_MEMORY_SCOPE_AGENT);
        } else {
            while (__hip_atomic_load(&ctrl[1], __ATOMIC_ACQUIRE,
                                     __HIP_MEMORY_SCOPE_AGENT) == gen) {
                __builtin_amdgcn_s_sleep(1);
            }
        }
    }
    __syncthreads();
}

// ---------------------------------------------------------------------------
// element-wise neuron update for one 16x16 C tile held as v8f (8 rows/lane)
// ---------------------------------------------------------------------------
__device__ __forceinline__ void neuron_update(v8f& acc, v8f& lifv, v8f& hy,
                                              v8f& hz, v8f& ref, v8f& cnt,
                                              float bias_n, int mbase, int g,
                                              int ncolabs, bf16* __restrict__ swrite) {
#pragma unroll
    for (int e = 0; e < 8; ++e) {
        float I  = acc[e] + bias_n;
        float v  = lifv[e] * LIF_KEEP + DT * I;          // LIF integrate
        float ls = v > 1.0f ? 1.0f : 0.0f;               // LIF spike
        lifv[e]  = v - ls;                               // soft reset
        float z  = hz[e] + DT * (SPIKE_GAIN * ls - GAMMA * hy[e] - EPSI * hz[e]);
        float y  = hy[e] + DT * z;
        float sp = (y - 1.0f - ref[e]) > 0.0f ? 1.0f : 0.0f;
        hy[e] = y;
        hz[e] = z;
        ref[e] = ref[e] * REF_DECAY + sp;
        cnt[e] += sp;
        int m = mbase + g * 8 + e;                       // C layout: M = v + 8*(lane/16)
        swrite[(size_t)m * N_HID + ncolabs] = (bf16)sp;
    }
}

// ---------------------------------------------------------------------------
// persistent recurrence kernel
// ---------------------------------------------------------------------------
__global__ void __launch_bounds__(256, 1)
esn_persistent_kernel(const bf16* __restrict__ xbf,    // [T][32][96]
                      const bf16* __restrict__ x2hbf,  // [96][2048]
                      const bf16* __restrict__ h2hbf,  // [2048][2048]
                      const float* __restrict__ bias,  // [2048]
                      bf16* __restrict__ sb0,          // [32][2048] ping
                      bf16* __restrict__ sb1,          // [32][2048] pong
                      float* __restrict__ out,         // [32][2048]
                      unsigned* ctrl) {
    __shared__ bf16 sA[BATCH * LDS_STRIDE];            // 131,584 B spike stage

    const int tid  = threadIdx.x;
    const int lane = tid & 31;
    const int wave = tid >> 5;
    const int g    = lane >> 4;     // half-wave group
    const int mrow = lane & 15;     // A-tile row
    const int ncol = lane & 15;     // C-tile column
    const int n0   = blockIdx.x * SLICE + wave * 16;   // this wave's 16 cols
    const int nabs = n0 + ncol;

    const float bias_n = bias[nabs];

    // persistent per-lane state: 2 M-tiles x 8 rows, fp32
    v8f lifv0 = {}, lifv1 = {}, hy0 = {}, hy1 = {}, hz0 = {}, hz1 = {};
    v8f ref0 = {}, ref1 = {}, cnt0 = {}, cnt1 = {};

    for (int t = 0; t < TSTEPS; ++t) {
        const bf16* sread  = (t & 1) ? sb1 : sb0;
        bf16*       swrite = (t & 1) ? sb0 : sb1;
        const bf16* xt     = xbf + (size_t)t * (BATCH * N_INP);

        // stage spike matrix into LDS (overlaps with x-projection below)
        issue_spike_copy(sread, sA, tid);

        v8f acc0 = {}, acc1 = {};

        // ---- input projection: xt[32x96] @ x2h[96 x n0..n0+16) ----
#pragma unroll
        for (int kt = 0; kt < 3; ++kt) {
            const int kb = kt * 32;
            v16bf b;
#pragma unroll
            for (int s = 0; s < 16; ++s) b[s] = x2hbf[(size_t)(kb + (s < 8 ? g * 8 + s : 16 + g * 8 + (s - 8))) * N_HID + n0 + ncol];
            v16bf a0 = load_a16(xt, N_INP, mrow,      kb, g);
            v16bf a1 = load_a16(xt, N_INP, 16 + mrow, kb, g);
            acc0 = __builtin_amdgcn_wmma_f32_16x16x32_bf16(false, a0, false, b,
                                                           (short)0, acc0, false, false);
            acc1 = __builtin_amdgcn_wmma_f32_16x16x32_bf16(false, a1, false, b,
                                                           (short)0, acc1, false, false);
        }

        wait_spike_copy();   // s_wait_asynccnt 0 + workgroup barrier

        // ---- recurrent projection: s[32x2048] @ h2h[2048 x n0..n0+16) ----
#pragma unroll 4
        for (int kt = 0; kt < 64; ++kt) {
            const int kb = kt * 32;
            v16bf b;
#pragma unroll
            for (int s = 0; s < 16; ++s) b[s] = h2hbf[(size_t)(kb + (s < 8 ? g * 8 + s : 16 + g * 8 + (s - 8))) * N_HID + n0 + ncol];
            v16bf a0 = load_a16(sA, LDS_STRIDE, mrow,      kb, g);
            v16bf a1 = load_a16(sA, LDS_STRIDE, 16 + mrow, kb, g);
            acc0 = __builtin_amdgcn_wmma_f32_16x16x32_bf16(false, a0, false, b,
                                                           (short)0, acc0, false, false);
            acc1 = __builtin_amdgcn_wmma_f32_16x16x32_bf16(false, a1, false, b,
                                                           (short)0, acc1, false, false);
        }

        // ---- neuron dynamics + spike emission ----
        neuron_update(acc0, lifv0, hy0, hz0, ref0, cnt0, bias_n,  0, g, nabs, swrite);
        neuron_update(acc1, lifv1, hy1, hz1, ref1, cnt1, bias_n, 16, g, nabs, swrite);

        // ---- device-wide step barrier ----
        grid_barrier(ctrl);
    }

    // ---- emit spike rates ----
    const float inv_t = 1.0f / (float)TSTEPS;
#pragma unroll
    for (int e = 0; e < 8; ++e) {
        int mloc = g * 8 + e;
        out[(size_t)mloc * N_HID + nabs]        = cnt0[e] * inv_t;
        out[(size_t)(16 + mloc) * N_HID + nabs] = cnt1[e] * inv_t;
    }
}

// ---------------------------------------------------------------------------
// host entry
// ---------------------------------------------------------------------------
extern "C" void kernel_launch(void* const* d_in, const int* in_sizes, int n_in,
                              void* d_out, int out_size, void* d_ws, size_t ws_size,
                              hipStream_t stream) {
    const float* x    = (const float*)d_in[0];   // [32,1024,96]
    const float* x2h  = (const float*)d_in[1];   // [96,2048]
    const float* h2h  = (const float*)d_in[2];   // [2048,2048]
    const float* bias = (const float*)d_in[3];   // [2048]
    float* out = (float*)d_out;

    // workspace carve-out (256B aligned)
    char* ws = (char*)d_ws;
    size_t off = 0;
    auto carve = [&](size_t bytes) -> char* {
        char* p = ws + off;
        off += (bytes + 255) & ~(size_t)255;
        return p;
    };
    bf16* h2hbf = (bf16*)carve((size_t)N_HID * N_HID * sizeof(bf16));      // 8 MB
    bf16* x2hbf = (bf16*)carve((size_t)N_INP * N_HID * sizeof(bf16));      // 384 KB
    bf16* xbf   = (bf16*)carve((size_t)TSTEPS * BATCH * N_INP * sizeof(bf16)); // 6 MB
    bf16* sb0   = (bf16*)carve((size_t)BATCH * N_HID * sizeof(bf16));      // 128 KB
    bf16* sb1   = (bf16*)carve((size_t)BATCH * N_HID * sizeof(bf16));      // 128 KB
    unsigned* ctrl = (unsigned*)carve(256);

    // per-launch reset (graph-capture safe, deterministic)
    (void)hipMemsetAsync(sb0, 0, (size_t)BATCH * N_HID * sizeof(bf16), stream);
    (void)hipMemsetAsync(sb1, 0, (size_t)BATCH * N_HID * sizeof(bf16), stream);
    (void)hipMemsetAsync(ctrl, 0, 256, stream);

    const int n_h2h = N_HID * N_HID;
    f32_to_bf16_kernel<<<(n_h2h + 255) / 256, 256, 0, stream>>>(h2h, h2hbf, n_h2h);
    const int n_x2h = N_INP * N_HID;
    f32_to_bf16_kernel<<<(n_x2h + 255) / 256, 256, 0, stream>>>(x2h, x2hbf, n_x2h);
    const int n_x = BATCH * TSTEPS * N_INP;
    x_time_major_bf16_kernel<<<(n_x + 255) / 256, 256, 0, stream>>>(x, xbf);

    esn_persistent_kernel<<<NBLOCKS, 256, 0, stream>>>(xbf, x2hbf, h2hbf, bias,
                                                       sb0, sb1, out, ctrl);
}
